// MemoryModule_3994319585532
// MI455X (gfx1250) — matrix operand order repaired and checked by reading, not compiled, not run on hardware
//
#include <hip/hip_runtime.h>
#include <hip/hip_bf16.h>

typedef __attribute__((ext_vector_type(16))) _Float16 v16h;
typedef __attribute__((ext_vector_type(8)))  _Float16 v8h;
typedef __attribute__((ext_vector_type(8)))  float    v8f;
typedef __attribute__((ext_vector_type(4)))  float    v4f;
typedef __attribute__((ext_vector_type(4)))  unsigned int u32x4;
typedef __attribute__((ext_vector_type(8)))  unsigned int u32x8;

#define B_    2
#define T_    4
#define CK_   64
#define CV_   128
#define HW_   576      // 24*24
#define HLWL_ 2304     // 48*48
#define NTOT_ 9216     // T*HLWL
#define OUTC_ 384
#define NITER_ 288     // NTOT_/32

__device__ __forceinline__ v8f wmma_f16(v16h a, v16h b, v8f c) {
  return __builtin_amdgcn_wmma_f32_16x16x32_f16(false, a, false, b, (short)0, c, false, false);
}

__device__ __forceinline__ v8f v8f_zero() {
  v8f z;
#pragma unroll
  for (int i = 0; i < 8; ++i) z[i] = 0.f;
  return z;
}

// ---------------------------------------------------------------------------
// TDM: one descriptor-driven DMA pulls a 32(n) x 128(cv) f32 tile of V into
// LDS, with hardware padding of 4 dwords every 32 dwords (144B row stride).
// D# group0: count=1 | lds_addr | global_addr | type=2
// D# group1: data_size=4B, pad_enable, pad_interval=32dw(code 4),
//            pad_amount=4dw(code 3), tensor_dim0=2304, tensor_dim1=128,
//            tile_dim0=32, tile_dim1=128, tensor_dim0_stride=2304
// ---------------------------------------------------------------------------
__device__ __forceinline__ void tdm_load_v(unsigned lds_off, const float* gptr) {
  const unsigned long long ga = (unsigned long long)(uintptr_t)gptr;
  u32x4 g0;
  g0[0] = 1u;                                   // count = 1 valid descriptor
  g0[1] = lds_off;                              // LDS byte address
  g0[2] = (unsigned)(ga & 0xFFFFFFFFu);         // global_addr[31:0]
  g0[3] = (unsigned)((ga >> 32) & 0x01FFFFFFu) | (2u << 30);  // addr[56:32] | type=2
  u32x8 g1;
  g1[0] = (2u << 16) | (1u << 20) | (4u << 22) | (3u << 25);  // dsz=4B, pad 32dw->+4dw
  g1[1] = (HLWL_ & 0xFFFFu) << 16;              // tensor_dim0[15:0]
  g1[2] = (CV_ & 0xFFFFu) << 16;                // tensor_dim0[31:16]=0 | tensor_dim1
  g1[3] = 32u << 16;                            // tensor_dim1[31:16]=0 | tile_dim0=32
  g1[4] = CV_;                                  // tile_dim1=128 | tile_dim2=0
  g1[5] = HLWL_;                                // tensor_dim0_stride[31:0]
  g1[6] = 0u;                                   // stride0[47:32]=0 | stride1[15:0]=0
  g1[7] = 0u;
  asm volatile("tensor_load_to_lds %0, %1" :: "s"(g0), "s"(g1) : "memory");
}

// ---------------------------------------------------------------------------
// Low branch: fused flash attention, double-buffered.
//   out[b, 256+cv, m] = sum_n V[cv,n] * softmax_n(0.125 * K[n,:]·Q[m,:])
// K chunk (32n x 64c) staged manually as f16 (needs transpose), packed
// ds_store_b128. V chunk (128cv x 32n) staged by TDM as f32. One barrier per
// step; wave 0 manages TENSORcnt.
// ---------------------------------------------------------------------------
__global__ void __launch_bounds__(128) low_attn_kernel(
    const float* __restrict__ mkl, const float* __restrict__ mvl,
    const float* __restrict__ qkl, float* __restrict__ out) {
  __shared__ __align__(16) _Float16 Ks[2][32][CK_ + 8];  // f16, rowstride 144B
  __shared__ __align__(16) float    Vs[2][CV_][36];      // f32, rowstride 144B (TDM pad)

  const int b      = blockIdx.x / 36;
  const int mBlock = (blockIdx.x % 36) * 64;
  const int tid    = threadIdx.x;
  const int wid    = tid >> 5;
  const int lane   = tid & 31;
  const int lhalf  = (lane < 16) ? 0 : 1;
  const int lm     = lane & 15;
  const int mBase  = mBlock + wid * 16;

  const float* kbase = mkl + (size_t)b * T_ * CK_ * HLWL_;
  const float* vbase = mvl + (size_t)b * T_ * CV_ * HLWL_;

  // Q as B-fragments. B layout: lane -> m, half j -> c = 32g + 16*lhalf + j
  v16h bq[2];
  {
    const float* qb = qkl + (size_t)b * CK_ * HLWL_ + (mBase + lm);
#pragma unroll
    for (int g = 0; g < 2; ++g)
#pragma unroll
      for (int j = 0; j < 16; ++j)
        bq[g][j] = (_Float16)qb[(size_t)(32 * g + 16 * lhalf + j) * HLWL_];
  }

  v8f acc[8];
#pragma unroll
  for (int v = 0; v < 8; ++v) acc[v] = v8f_zero();
  float m_run = -1e30f, l_run = 0.f;

  // K staging: thread owns (n_local = tid&31, 16 c values), packed b128 stores.
  auto stageK = [&](int it, int buf) {
    const int nBase = it * 32;
    const int t = nBase / HLWL_, hw = nBase % HLWL_;   // 2304%32==0: no straddle
    const float* kg = kbase + (size_t)t * CK_ * HLWL_ + hw;
    const int nl = tid & 31, c0 = (tid >> 5) * 16;
    v8h p0, p1;
#pragma unroll
    for (int i = 0; i < 8; ++i) p0[i] = (_Float16)kg[(size_t)(c0 + i) * HLWL_ + nl];
#pragma unroll
    for (int i = 0; i < 8; ++i) p1[i] = (_Float16)kg[(size_t)(c0 + 8 + i) * HLWL_ + nl];
    *(v8h*)&Ks[buf][nl][c0] = p0;
    *(v8h*)&Ks[buf][nl][c0 + 8] = p1;
    __builtin_prefetch(kg + (size_t)c0 * HLWL_ + nl + 64, 0, 0);  // 2 chunks ahead
  };
  auto issueV = [&](int it, int buf) {
    const int nBase = it * 32;
    const int t = nBase / HLWL_, hw = nBase % HLWL_;
    tdm_load_v((unsigned)(uintptr_t)&Vs[buf][0][0],
               vbase + (size_t)t * CV_ * HLWL_ + hw);
  };

  stageK(0, 0);
  if (wid == 0) issueV(0, 0);

  for (int it = 0; it < NITER_; ++it) {
    if (wid == 0) __builtin_amdgcn_s_wait_tensorcnt(0);  // TDM(it) done
    __syncthreads();                                     // K(it) visible, bufs free
    if (it + 1 < NITER_) {
      stageK(it + 1, (it + 1) & 1);
      if (wid == 0) issueV(it + 1, (it + 1) & 1);
    }
    const int kb = it & 1;

    // S tiles: d=0 -> keys n=0..15, d=1 -> 16..31 (rows); m in lanes.
    v8f s[2];
#pragma unroll
    for (int d = 0; d < 2; ++d) {
      v8f c = v8f_zero();
      const int n = 16 * d + lm;
#pragma unroll
      for (int g = 0; g < 2; ++g) {
        v8h k0 = *(const v8h*)&Ks[kb][n][32 * g + 16 * lhalf];
        v8h k1 = *(const v8h*)&Ks[kb][n][32 * g + 16 * lhalf + 8];
        v16h a = __builtin_shufflevector(k0, k1, 0, 1, 2, 3, 4, 5, 6, 7,
                                         8, 9, 10, 11, 12, 13, 14, 15);
        c = wmma_f16(a, bq[g], c);
      }
      s[d] = c;
    }

    // Online softmax over n per m column (rows split between lane and lane^16).
    float cmax = -1e30f;
#pragma unroll
    for (int r = 0; r < 8; ++r) {
      s[0][r] *= 0.125f; s[1][r] *= 0.125f;
      cmax = fmaxf(cmax, fmaxf(s[0][r], s[1][r]));
    }
    cmax = fmaxf(cmax, __shfl_xor(cmax, 16, 32));
    const float m_new = fmaxf(m_run, cmax);
    const float corr  = __expf(m_run - m_new);
    float e0[8], e1[8], rsum = 0.f;
#pragma unroll
    for (int r = 0; r < 8; ++r) {
      e0[r] = __expf(s[0][r] - m_new);
      e1[r] = __expf(s[1][r] - m_new);
      rsum += e0[r] + e1[r];
    }
    rsum += __shfl_xor(rsum, 16, 32);
    l_run = l_run * corr + rsum;
    m_run = m_new;
#pragma unroll
    for (int v = 0; v < 8; ++v)
#pragma unroll
      for (int r = 0; r < 8; ++r) acc[v][r] *= corr;

    // Repack P (32n x 16m) into a B-fragment in registers via one lane^16 swap.
    v16h bp;
#pragma unroll
    for (int r = 0; r < 8; ++r) {
      float p0 = __shfl_xor(e0[r], 16, 32);
      float p1 = __shfl_xor(e1[r], 16, 32);
      bp[r]     = (_Float16)((lane < 16) ? e0[r] : p1);
      bp[r + 8] = (_Float16)((lane < 16) ? p0 : e1[r]);
    }

    // O[cv, m] += V_tile(16cv x 32n) x P(32n x 16m); V converted f32->f16 here.
#pragma unroll
    for (int vt = 0; vt < 8; ++vt) {
      const int cv = 16 * vt + lm;
      const float* vr = &Vs[kb][cv][16 * lhalf];
      v16h av;
#pragma unroll
      for (int grp = 0; grp < 2; ++grp) {
        v4f x0 = *(const v4f*)(vr + 8 * grp);
        v4f x1 = *(const v4f*)(vr + 8 * grp + 4);
#pragma unroll
        for (int w = 0; w < 4; ++w) {
          av[8 * grp + w]     = (_Float16)x0[w];
          av[8 * grp + 4 + w] = (_Float16)x1[w];
        }
      }
      acc[vt] = wmma_f16(av, bp, acc[vt]);
    }
  }

  const float inv_l = 1.f / l_run;
  const int m = mBase + lm;
#pragma unroll
  for (int vt = 0; vt < 8; ++vt)
#pragma unroll
    for (int r = 0; r < 8; ++r) {
      int cv = 16 * vt + 8 * lhalf + r;
      out[((size_t)b * OUTC_ + 256 + cv) * HLWL_ + m] = acc[vt][r] * inv_l;
    }
}

// ---------------------------------------------------------------------------
// High branch stage 1: g_attn_raw[b,t,k,v] = (1/sqrt(128)) * sum_n gk*gv
// ---------------------------------------------------------------------------
__global__ void __launch_bounds__(128) h1_gattn_kernel(
    const float* __restrict__ mk, const float* __restrict__ mv,
    float* __restrict__ gattn) {
  const int tid = threadIdx.x, wid = tid >> 5, lane = tid & 31;
  const int lhalf = (lane < 16) ? 0 : 1, lm = lane & 15;
  const int tile = blockIdx.x * 4 + wid;        // 0..255
  const int vt = tile & 7, kt = (tile >> 3) & 3;
  const int t = (tile >> 5) & 3, b = tile >> 7;

  const float* gk = mk + (((size_t)b * T_ + t) * CK_ + (kt * 16 + lm)) * HW_;
  const float* gv = mv + (((size_t)b * T_ + t) * CV_ + (vt * 16 + lm)) * HW_;

  v8f acc = v8f_zero();
  for (int nc = 0; nc < HW_; nc += 32) {
    v16h a, bb;
#pragma unroll
    for (int grp = 0; grp < 2; ++grp)
#pragma unroll
      for (int w = 0; w < 8; ++w)
        a[8 * grp + w] = (_Float16)gk[nc + 16 * lhalf + 8 * grp + w];
#pragma unroll
    for (int j = 0; j < 16; ++j)
      bb[j] = (_Float16)gv[nc + 16 * lhalf + j];
    acc = wmma_f16(a, bb, acc);
  }
  const float scale = 0.08838834764831845f;  // 128^-0.5
  const int k = kt * 16 + 8 * lhalf;
#pragma unroll
  for (int r = 0; r < 8; ++r)
    gattn[(((size_t)b * T_ + t) * CK_ + (k + r)) * CV_ + (vt * 16 + lm)] = acc[r] * scale;
}

// Stage 2: softmax over T (4 frames) in place.
__global__ void h2_softmax_t_kernel(float* __restrict__ gattn) {
  const int idx = blockIdx.x * blockDim.x + threadIdx.x;  // 0..16383
  if (idx >= B_ * CK_ * CV_) return;
  const int b = idx >> 13, kv = idx & 8191;
  float* p = gattn + (size_t)b * T_ * CK_ * CV_ + kv;
  float x0 = p[0], x1 = p[8192], x2 = p[16384], x3 = p[24576];
  float mx = fmaxf(fmaxf(x0, x1), fmaxf(x2, x3));
  x0 = __expf(x0 - mx); x1 = __expf(x1 - mx);
  x2 = __expf(x2 - mx); x3 = __expf(x3 - mx);
  float inv = 1.f / (x0 + x1 + x2 + x3);
  p[0] = x0 * inv; p[8192] = x1 * inv; p[16384] = x2 * inv; p[24576] = x3 * inv;
}

// Stage 3: qsmall[b, t*64+k, n'] = sum_v gattn[b,t,k,v] * qv[b,v,n']
__global__ void __launch_bounds__(128) h3_qout_kernel(
    const float* __restrict__ gattn, const float* __restrict__ qv,
    float* __restrict__ qsmall) {
  const int tid = threadIdx.x, wid = tid >> 5, lane = tid & 31;
  const int lhalf = (lane < 16) ? 0 : 1, lm = lane & 15;
  const int tile = blockIdx.x * 4 + wid;        // 0..1151
  const int nt = tile % 36;
  const int rest = tile / 36;                    // 0..31
  const int kt = rest & 3, t = (rest >> 2) & 3, b = rest >> 4;

  const float* ga = gattn + (((size_t)b * T_ + t) * CK_ + (kt * 16 + lm)) * CV_;
  const float* qb = qv + (size_t)b * CV_ * HW_ + (nt * 16 + lm);

  v8f acc = v8f_zero();
  for (int vc = 0; vc < CV_; vc += 32) {
    v16h a, bb;
#pragma unroll
    for (int grp = 0; grp < 2; ++grp)
#pragma unroll
      for (int w = 0; w < 8; ++w)
        a[8 * grp + w] = (_Float16)ga[vc + 16 * lhalf + 8 * grp + w];
#pragma unroll
    for (int j = 0; j < 16; ++j)
      bb[j] = (_Float16)qb[(size_t)(vc + 16 * lhalf + j) * HW_];
    acc = wmma_f16(a, bb, acc);
  }
  const int k = kt * 16 + 8 * lhalf;
#pragma unroll
  for (int r = 0; r < 8; ++r)
    qsmall[((size_t)b * 256 + t * 64 + (k + r)) * HW_ + (nt * 16 + lm)] = acc[r];
}

// Stage 4: half-pixel bilinear 24x24 -> 48x48 into out channels 0..255.
__global__ void h4_upsample_kernel(const float* __restrict__ qsmall,
                                   float* __restrict__ out) {
  const int idx = blockIdx.x * blockDim.x + threadIdx.x;
  const int total = B_ * 256 * HLWL_;
  if (idx >= total) return;
  const int x = idx % 48, y = (idx / 48) % 48;
  const int ch = (idx / HLWL_) % 256, b = idx / (HLWL_ * 256);

  int y0 = (y - 1) >> 1, x0 = (x - 1) >> 1;
  const float wy0 = (y & 1) ? 0.75f : 0.25f, wy1 = 1.f - wy0;
  const float wx0 = (x & 1) ? 0.75f : 0.25f, wx1 = 1.f - wx0;
  int y1 = y0 + 1, x1 = x0 + 1;
  y0 = max(y0, 0); x0 = max(x0, 0);
  y1 = min(y1, 23); x1 = min(x1, 23);

  const float* src = qsmall + ((size_t)b * 256 + ch) * HW_;
  float v = wy0 * (wx0 * src[y0 * 24 + x0] + wx1 * src[y0 * 24 + x1]) +
            wy1 * (wx0 * src[y1 * 24 + x0] + wx1 * src[y1 * 24 + x1]);
  out[((size_t)b * OUTC_ + ch) * HLWL_ + y * 48 + x] = v;
}

extern "C" void kernel_launch(void* const* d_in, const int* in_sizes, int n_in,
                              void* d_out, int out_size, void* d_ws, size_t ws_size,
                              hipStream_t stream) {
  const float* mk  = (const float*)d_in[0];  // memory_keys      [2,4,64,24,24]
  const float* mv  = (const float*)d_in[1];  // memory_values    [2,4,128,24,24]
  const float* qv  = (const float*)d_in[2];  // query_value      [2,128,24,24]
  const float* mkl = (const float*)d_in[3];  // memory_keys_low  [2,4,64,48,48]
  const float* mvl = (const float*)d_in[4];  // memory_values_low[2,4,128,48,48]
  const float* qkl = (const float*)d_in[5];  // query_key_low    [2,64,48,48]
  float* out = (float*)d_out;

  float* gattn  = (float*)d_ws;              // 2*4*64*128 = 65536 f32
  float* qsmall = (float*)d_ws + 65536;      // 2*256*576  = 294912 f32

  low_attn_kernel<<<72, 128, 0, stream>>>(mkl, mvl, qkl, out);
  h1_gattn_kernel<<<64, 128, 0, stream>>>(mk, mv, gattn);
  h2_softmax_t_kernel<<<64, 256, 0, stream>>>(gattn);
  h3_qout_kernel<<<288, 128, 0, stream>>>(gattn, qv, qsmall);
  h4_upsample_kernel<<<(B_ * 256 * HLWL_ + 255) / 256, 256, 0, stream>>>(qsmall, out);
}
